// ImprovedProbabilisticRadarBranch_1494648619444
// MI455X (gfx1250) — matrix operand order repaired and checked by reading, not compiled, not run on hardware
//
#include <hip/hip_runtime.h>
#include <hip/hip_bf16.h>

typedef __attribute__((ext_vector_type(16))) _Float16 v16h;
typedef __attribute__((ext_vector_type(8)))  float    v8f;

#define BATCH  2
#define NGAUSS 2048
#define CCH    256
#define HDIM   64
#define WDIM   176
#define NPIX   (HDIM * WDIM)   // 11264
#define NTILE  (NPIX / 16)     // 704
#define CEXT   272             // 256 feats + 1 (density) + 1 (avg_sigma) + 14 zero pad
#define ROWS   (BATCH * NGAUSS)

// ---------------------------------------------------------------------------
// Generic row-major linear layer: out[r][j] = (relu)(bias[j] + sum_k in[r][k]*w[k][j])
// ---------------------------------------------------------------------------
__global__ void linear_kernel(const float* __restrict__ in, const float* __restrict__ w,
                              const float* __restrict__ bias, float* __restrict__ out,
                              int K, int Cout, int total, int do_relu) {
    int id = blockIdx.x * blockDim.x + threadIdx.x;
    if (id >= total) return;
    int r = id / Cout;
    int j = id - r * Cout;
    float s = bias[j];
    const float* ir = in + (size_t)r * K;
    for (int k = 0; k < K; ++k) s = fmaf(ir[k], w[(size_t)k * Cout + j], s);
    out[id] = do_relu ? fmaxf(s, 0.0f) : s;
}

// ---------------------------------------------------------------------------
// Final feature transform layer (ft_w2, no relu): writes f16 output transposed
// into the WMMA A-panel layout: featsA[(b*CEXT + c)*NGAUSS + n]
// ---------------------------------------------------------------------------
__global__ void final_linear_f16_kernel(const float* __restrict__ in, const float* __restrict__ w,
                                        const float* __restrict__ bias,
                                        _Float16* __restrict__ featsA) {
    int id = blockIdx.x * blockDim.x + threadIdx.x;
    if (id >= ROWS * CCH) return;
    int r = id >> 8;        // row (b*NGAUSS + n)
    int j = id & 255;       // output channel
    int b = r / NGAUSS;
    int n = r - b * NGAUSS;
    float s = bias[j];
    const float* ir = in + (size_t)r * CCH;
    for (int k = 0; k < CCH; ++k) s = fmaf(ir[k], w[(size_t)k * CCH + j], s);
    featsA[((size_t)b * CEXT + j) * NGAUSS + n] = (_Float16)s;
}

// ---------------------------------------------------------------------------
// Per-gaussian projection/mask params + fill the two extra A-panel channels:
// channel 256 = 1.0 (density row), channel 257 = avg_sigma, 258..271 = 0
// ---------------------------------------------------------------------------
__global__ void param_kernel(const float* __restrict__ g, const float* __restrict__ intr,
                             float* __restrict__ px, float* __restrict__ py,
                             float* __restrict__ sxi, float* __restrict__ syi,
                             float* __restrict__ wf, _Float16* __restrict__ featsA) {
    int id = blockIdx.x * blockDim.x + threadIdx.x;
    if (id >= ROWS) return;
    int b = id / NGAUSS;
    int n = id - b * NGAUSS;
    const float* gd = g + (size_t)id * 14;
    float x = gd[0], y = gd[1], z = gd[2];
    float fx = intr[0], cx = intr[2], fy = intr[4], cy = intr[5];
    float sX = (float)WDIM / cx * 0.5f;
    float sY = (float)HDIM / cy * 0.5f;
    float invz = 1.0f / (z + 1e-6f);
    float pxv = (fx * x + cx * z) * invz * sX;
    float pyv = (fy * y + cy * z) * invz * sY;
    bool valid = z > 0.1f;
    bool inb = (pxv >= 0.0f) && (pxv < (float)WDIM) && (pyv >= 0.0f) && (pyv < (float)HDIM);
    float sx = fmaxf(gd[5] * sX, 1.0f);
    float sy = fmaxf(gd[6] * sY, 1.0f);
    float weff = (valid && inb) ? gd[12] : 0.0f;
    px[id] = pxv; py[id] = pyv;
    sxi[id] = 1.0f / sx; syi[id] = 1.0f / sy;
    wf[id] = weff;
    float avg = 0.5f * (sx + sy);
    size_t base = (size_t)b * CEXT * NGAUSS;
    featsA[base + (size_t)256 * NGAUSS + n] = (_Float16)1.0f;
    featsA[base + (size_t)257 * NGAUSS + n] = (_Float16)avg;
    for (int c = 258; c < CEXT; ++c) featsA[base + (size_t)c * NGAUSS + n] = (_Float16)0.0f;
}

// ---------------------------------------------------------------------------
// Splat kernel: per block, one batch x one 16-pixel tile. 17 waves, each wave
// owns a 16-channel tile. K loop processes 64 gaussians per iteration:
// waves 0..15 build two 32x16 f16 gw B-tiles in LDS (coalesced param loads,
// one exp per thread per sub-tile), then every wave issues two
// v_wmma_f32_16x16x32_f16. Wave 16's channels are density/uncertainty.
// ---------------------------------------------------------------------------
__global__ __launch_bounds__(544) void splat_kernel(
    const _Float16* __restrict__ featsA,
    const float* __restrict__ pxv, const float* __restrict__ pyv,
    const float* __restrict__ sxiv, const float* __restrict__ syiv,
    const float* __restrict__ wv,
    float* __restrict__ out) {
    int blk = blockIdx.x;
    int b = blk / NTILE;
    int tile = blk - b * NTILE;
    int pixbase = tile * 16;
    int t = threadIdx.x;
    int wave = t >> 5;
    int lane = t & 31;

    __shared__ __attribute__((aligned(32))) unsigned int ldsB[512]; // two 32x16 f16 B tiles
    __shared__ float ldsDen[16];
    __shared__ float ldsUnc[16];

    // gw writer identity: waves 0..15 compute pixel n=wave, gaussians k=lane, lane+32
    float xf = 0.0f, yf = 0.0f;
    int hIdx = 0;
    if (wave < 16) {
        int p = pixbase + wave;
        xf = (float)(p % WDIM);
        yf = (float)(p / WDIM);
        // B layout: lanes 0-15 hold K={0..7,16..23}, lanes 16-31 hold K={8..15,24..31}; N = lane&15
        hIdx = (wave + ((lane >> 3) & 1) * 16) * 16 + (lane & 7) + ((lane >> 4) << 3);
    }

    // A fragment identity (16-bit A 16x32 layout):
    // lanes 0-15: M=lane, K pairs {0..7,16..23}; lanes 16-31: M=lane-16, K pairs {8..15,24..31}
    int m = lane & 15;
    int ch = wave * 16 + m;
    int klane = (lane >> 4) * 8;
    const unsigned int* fw = (const unsigned int*)featsA;
    size_t rowWordBase = ((size_t)b * CEXT + ch) * (NGAUSS / 2);
    int gbase = b * NGAUSS;

    _Float16* hb = (_Float16*)ldsB;
    v8f acc = {};

    for (int kb = 0; kb < NGAUSS; kb += 64) {
        __syncthreads(); // previous B tiles fully consumed
        if (wave < 16) {
            int gi = gbase + kb + lane;      // coalesced: lane == gaussian index
            // sub-tile 0: gaussians kb+lane
            {
                float dx = (xf - pxv[gi]) * sxiv[gi];
                float dy = (yf - pyv[gi]) * syiv[gi];
                float d = dx * dx + dy * dy;
                float e = __expf(-0.5f * d) * wv[gi];
                hb[hIdx] = (_Float16)((d < 9.0f) ? e : 0.0f);
            }
            // sub-tile 1: gaussians kb+32+lane
            {
                int gj = gi + 32;
                float dx = (xf - pxv[gj]) * sxiv[gj];
                float dy = (yf - pyv[gj]) * syiv[gj];
                float d = dx * dx + dy * dy;
                float e = __expf(-0.5f * d) * wv[gj];
                hb[hIdx + 512] = (_Float16)((d < 9.0f) ? e : 0.0f);
            }
        }
        __syncthreads();
        union { unsigned int u[8]; v16h v; } a0, a1;
        size_t wb = rowWordBase + (size_t)((kb + klane) >> 1);
        #pragma unroll
        for (int i = 0; i < 4; ++i) a0.u[i] = fw[wb + i];
        #pragma unroll
        for (int i = 0; i < 4; ++i) a0.u[4 + i] = fw[wb + 8 + i];
        #pragma unroll
        for (int i = 0; i < 4; ++i) a1.u[i] = fw[wb + 16 + i];
        #pragma unroll
        for (int i = 0; i < 4; ++i) a1.u[4 + i] = fw[wb + 24 + i];
        v16h bv0 = *(const v16h*)(&ldsB[lane * 8]);
        v16h bv1 = *(const v16h*)(&ldsB[256 + lane * 8]);
        acc = __builtin_amdgcn_wmma_f32_16x16x32_f16(
            false, a0.v, false, bv0, (short)0, acc, false, false);
        acc = __builtin_amdgcn_wmma_f32_16x16x32_f16(
            false, a1.v, false, bv1, (short)0, acc, false, false);
    }

    // wave 16: D row M=0 = density, M=1 = uncertainty (lanes 0-15)
    if (wave == 16 && lane < 16) { ldsDen[lane] = acc[0]; ldsUnc[lane] = acc[1]; }
    __syncthreads();

    int n = lane & 15;
    float den = fmaxf(ldsDen[n], 1e-6f);
    int p = pixbase + n;
    if (wave < 16) {
        #pragma unroll
        for (int r = 0; r < 8; ++r) {
            int c = wave * 16 + r + ((lane >> 4) << 3); // M = r (lanes<16) or r+8
            out[((size_t)b * CCH + c) * NPIX + p] = acc[r] / den;
        }
    } else if (lane < 16) {
        size_t uoff = (size_t)BATCH * CCH * NPIX;
        out[uoff + (size_t)b * NPIX + p] = ldsUnc[n] / den;
        out[uoff + (size_t)BATCH * NPIX + (size_t)b * NPIX + p] = den;
    }
}

// ---------------------------------------------------------------------------
extern "C" void kernel_launch(void* const* d_in, const int* in_sizes, int n_in,
                              void* d_out, int out_size, void* d_ws, size_t ws_size,
                              hipStream_t stream) {
    (void)in_sizes; (void)n_in; (void)out_size; (void)ws_size;

    const float* g    = (const float*)d_in[0];
    const float* intr = (const float*)d_in[1];
    const float* w1 = (const float*)d_in[2];  const float* b1 = (const float*)d_in[3];
    const float* w2 = (const float*)d_in[4];  const float* b2 = (const float*)d_in[5];
    const float* w3 = (const float*)d_in[6];  const float* b3 = (const float*)d_in[7];
    const float* fw1 = (const float*)d_in[8]; const float* fb1 = (const float*)d_in[9];
    const float* fw2 = (const float*)d_in[10]; const float* fb2 = (const float*)d_in[11];
    float* out = (float*)d_out;

    float* ws    = (float*)d_ws;
    float* h1    = ws;                           // ROWS*64
    float* h2    = h1 + (size_t)ROWS * 64;       // ROWS*128
    float* feats = h2 + (size_t)ROWS * 128;      // ROWS*256
    float* ft1   = feats + (size_t)ROWS * 256;   // ROWS*256
    float* px    = ft1 + (size_t)ROWS * 256;
    float* py    = px + ROWS;
    float* sxi   = py + ROWS;
    float* syi   = sxi + ROWS;
    float* wf    = syi + ROWS;
    _Float16* featsA = (_Float16*)(wf + ROWS);   // BATCH*CEXT*NGAUSS halves

    linear_kernel<<<(ROWS * 64 + 255) / 256, 256, 0, stream>>>(g, w1, b1, h1, 14, 64, ROWS * 64, 1);
    linear_kernel<<<(ROWS * 128 + 255) / 256, 256, 0, stream>>>(h1, w2, b2, h2, 64, 128, ROWS * 128, 1);
    linear_kernel<<<(ROWS * 256 + 255) / 256, 256, 0, stream>>>(h2, w3, b3, feats, 128, 256, ROWS * 256, 0);
    linear_kernel<<<(ROWS * 256 + 255) / 256, 256, 0, stream>>>(feats, fw1, fb1, ft1, 256, 256, ROWS * 256, 1);
    final_linear_f16_kernel<<<(ROWS * 256 + 255) / 256, 256, 0, stream>>>(ft1, fw2, fb2, featsA);
    param_kernel<<<(ROWS + 255) / 256, 256, 0, stream>>>(g, intr, px, py, sxi, syi, wf, featsA);
    splat_kernel<<<BATCH * NTILE, 544, 0, stream>>>(featsA, px, py, sxi, syi, wf, out);
}
